// GRU_68066641707055
// MI455X (gfx1250) — matrix operand order repaired and checked by reading, not compiled, not run on hardware
//
#include <hip/hip_runtime.h>
#include <hip/hip_bf16.h>

#define VOCAB  4096
#define HIDDEN 1024
#define BATCH  64
#define SEQ    256
#define NWG    64          // workgroups in persistent scan kernel (512 waves)

typedef __attribute__((ext_vector_type(16))) _Float16 v16h;
typedef __attribute__((ext_vector_type(8)))  _Float16 v8h;
typedef __attribute__((ext_vector_type(8)))  float    v8f;

union ABFrag { v16h v; v8h h[2]; };

// ---------------------------------------------------------------------------
// WMMA fragment loaders (wave32, v_wmma_f32_16x16x32_f16 operand layouts).
__device__ inline v16h load_a_frag(const _Float16* __restrict__ A, int lda,
                                   int row0, int kc, int lane) {
  int m  = row0 + (lane & 15);
  int ko = (lane & 16) ? 8 : 0;
  ABFrag u;
  u.h[0] = *(const v8h*)(A + (size_t)m * lda + kc + ko);
  u.h[1] = *(const v8h*)(A + (size_t)m * lda + kc + 16 + ko);
  return u.v;
}

__device__ inline v16h load_b_frag(const _Float16* __restrict__ Bt, int ldb,
                                   int col0, int kc, int lane) {
  int n  = col0 + (lane & 15);
  int kb = kc + ((lane & 16) ? 8 : 0);
  ABFrag u;
  u.h[0] = *(const v8h*)(Bt + (size_t)n * ldb + kb);
  u.h[1] = *(const v8h*)(Bt + (size_t)n * ldb + kb + 16);
  return u.v;
}

// Fast activations: v_exp_f32 + v_rcp_f32, no IEEE division / libcalls.
__device__ inline float fast_rcp(float x) { return __builtin_amdgcn_rcpf(x); }
__device__ inline float sigmoid_f(float x) { return fast_rcp(1.0f + __expf(-x)); }
__device__ inline float tanh_f(float x) {
  float ax = __builtin_fabsf(x);
  float e  = __expf(-2.0f * ax);
  float y  = (1.0f - e) * fast_rcp(1.0f + e);
  return __builtin_copysignf(y, x);
}

// ---------------------------------------------------------------------------
// Grid-wide split barrier (generation counting, agent scope).
__device__ inline void grid_sync(unsigned* cnt, unsigned* gen, unsigned nwg) {
  __syncthreads();
  if (threadIdx.x == 0) {
    __threadfence();
    unsigned g = __hip_atomic_load(gen, __ATOMIC_SEQ_CST, __HIP_MEMORY_SCOPE_AGENT);
    unsigned a = __hip_atomic_fetch_add(cnt, 1u, __ATOMIC_SEQ_CST, __HIP_MEMORY_SCOPE_AGENT);
    if (a == nwg - 1u) {
      __hip_atomic_store(cnt, 0u, __ATOMIC_SEQ_CST, __HIP_MEMORY_SCOPE_AGENT);
      __hip_atomic_fetch_add(gen, 1u, __ATOMIC_SEQ_CST, __HIP_MEMORY_SCOPE_AGENT);
    } else {
      while (__hip_atomic_load(gen, __ATOMIC_SEQ_CST, __HIP_MEMORY_SCOPE_AGENT) == g)
        __builtin_amdgcn_s_sleep(2);
    }
    __threadfence();
  }
  __syncthreads();
}

// ---------------------------------------------------------------------------
// One-time converters: fp32 weights -> transposed f16 (B operand layout).
__global__ void convert_transpose_kernel(const float* __restrict__ src,
                                         _Float16* __restrict__ dst,
                                         int K, int N, int row_off, int src_ld) {
  size_t i = (size_t)blockIdx.x * blockDim.x + threadIdx.x;
  if (i >= (size_t)K * N) return;
  int n = (int)(i % N);
  int k = (int)(i / N);
  dst[(size_t)n * K + k] = (_Float16)src[(size_t)(k + row_off) * src_ld + n];
}

__global__ void init_kernel(float* __restrict__ Hf32, _Float16* __restrict__ Hzero,
                            unsigned* __restrict__ bar) {
  int i = blockIdx.x * blockDim.x + threadIdx.x;
  if (i < BATCH * HIDDEN) { Hf32[i] = 0.0f; Hzero[i] = (_Float16)0.0f; }
  if (i < 2) bar[i] = 0u;
}

// ---------------------------------------------------------------------------
// Persistent GRU scan: all 256 timesteps in one launch, 64 WGs x 256 thr.
//  Phase 1 (512 waves):  z & r gate tiles (1 tile/wave, K=1024).
//  Phase 2 (512 waves):  candidate tiles, 2 waves/tile K-split (K=512 each),
//                        LDS reduction inside the WG, then state update.
__global__ void __launch_bounds__(256)
gru_scan_kernel(const _Float16* __restrict__ Wzt,
                const _Float16* __restrict__ Wrt,
                const _Float16* __restrict__ Wht,
                const float* __restrict__ Wz,
                const float* __restrict__ Wr,
                const float* __restrict__ Wh,
                const float* __restrict__ bz,
                const float* __restrict__ br,
                const float* __restrict__ bh,
                const int*   __restrict__ X,
                const _Float16* __restrict__ Hzero,
                float* __restrict__ Hf32,
                float* __restrict__ zbuf,
                _Float16* __restrict__ rH,
                _Float16* __restrict__ Hall,
                unsigned* __restrict__ bar)
{
  const int lane = threadIdx.x & 31;
  const int wIn  = (int)(threadIdx.x >> 5);                             // 0..7
  const int wave = (int)((blockIdx.x * blockDim.x + threadIdx.x) >> 5); // 0..511

  // ---- Phase-1 tile assignment (fixed across timesteps) ----
  const int gate   = wave >> 8;
  const int g_row0 = ((wave >> 6) & 3) * 16;
  const int g_col0 = (wave & 63) * 16;
  const int g_n    = g_col0 + (lane & 15);
  const int g_mb   = g_row0 + ((lane & 16) ? 8 : 0);
  const _Float16* g_Bt  = gate ? Wrt : Wzt;
  const float*    g_emb = gate ? Wr  : Wz;
  const float     g_bn  = (gate ? br : bz)[g_n];

  // ---- Phase-2 assignment: WG hosts 4 tiles, 2 waves per tile (K-split) ----
  const int c_tile  = blockIdx.x * 4 + (wIn & 3);   // 0..255
  const int c_khalf = wIn >> 2;                     // 0 or 1
  const int c_row0  = (c_tile >> 6) * 16;
  const int c_col0  = (c_tile & 63) * 16;
  const int c_n     = c_col0 + (lane & 15);
  const int c_mb    = c_row0 + ((lane & 16) ? 8 : 0);
  const float c_bn  = bh[c_n];

  __shared__ float red[4][32][8];   // partial-K accumulators, 4 KB

  unsigned* bcnt = bar;
  unsigned* bgen = bar + 1;

#pragma unroll 1
  for (int t = 0; t < SEQ; ++t) {
    const _Float16* Hprev =
        (t == 0) ? Hzero : (Hall + (size_t)(t - 1) * BATCH * HIDDEN);

    // ==== Phase 1: z and r gates ==========================================
    {
      // Prefetch epilogue operands so their latency hides under the K-loop.
      int toks[8];
#pragma unroll
      for (int v = 0; v < 8; ++v) toks[v] = X[(g_mb + v) * SEQ + t];
      float emb[8];
#pragma unroll
      for (int v = 0; v < 8; ++v)
        emb[v] = g_emb[(size_t)toks[v] * HIDDEN + g_n];
      float hold[8];
      if (gate) {
#pragma unroll
        for (int v = 0; v < 8; ++v) hold[v] = Hf32[(g_mb + v) * HIDDEN + g_n];
      }

      v8f acc = {};
#pragma unroll 4
      for (int kc = 0; kc < HIDDEN; kc += 32) {
        v16h a = load_a_frag(Hprev, HIDDEN, g_row0, kc, lane);
        v16h b = load_b_frag(g_Bt, HIDDEN, g_col0, kc, lane);
        acc = __builtin_amdgcn_wmma_f32_16x16x32_f16(false, a, false, b,
                                                     (short)0, acc, false, false);
      }

      if (gate == 0) {
#pragma unroll
        for (int v = 0; v < 8; ++v) {
          float gz = sigmoid_f(acc[v] + emb[v] + g_bn);
          zbuf[(g_mb + v) * HIDDEN + g_n] = gz;
        }
      } else {
#pragma unroll
        for (int v = 0; v < 8; ++v) {
          float gr = sigmoid_f(acc[v] + emb[v] + g_bn);
          rH[(g_mb + v) * HIDDEN + g_n] = (_Float16)(gr * hold[v]);
        }
      }
    }
    grid_sync(bcnt, bgen, NWG);

    // ==== Phase 2: candidate + state update (K-split across wave pairs) ===
    {
      // Prefetch epilogue operands (khalf==0 waves do the epilogue).
      float emb2[8], zb[8], hold2[8];
      if (c_khalf == 0) {
        int toks2[8];
#pragma unroll
        for (int v = 0; v < 8; ++v) toks2[v] = X[(c_mb + v) * SEQ + t];
#pragma unroll
        for (int v = 0; v < 8; ++v)
          emb2[v] = Wh[(size_t)toks2[v] * HIDDEN + c_n];
#pragma unroll
        for (int v = 0; v < 8; ++v) zb[v] = zbuf[(c_mb + v) * HIDDEN + c_n];
#pragma unroll
        for (int v = 0; v < 8; ++v) hold2[v] = Hf32[(c_mb + v) * HIDDEN + c_n];
      }

      v8f acc = {};
      const int k0 = c_khalf * (HIDDEN / 2);
#pragma unroll 4
      for (int kc = k0; kc < k0 + HIDDEN / 2; kc += 32) {
        v16h a = load_a_frag(rH, HIDDEN, c_row0, kc, lane);
        v16h b = load_b_frag(Wht, HIDDEN, c_col0, kc, lane);
        acc = __builtin_amdgcn_wmma_f32_16x16x32_f16(false, a, false, b,
                                                     (short)0, acc, false, false);
      }

      if (c_khalf == 1) {
#pragma unroll
        for (int v = 0; v < 8; ++v) red[wIn & 3][lane][v] = acc[v];
      }
      __syncthreads();
      if (c_khalf == 0) {
        _Float16* Hout = Hall + (size_t)t * BATCH * HIDDEN;
#pragma unroll
        for (int v = 0; v < 8; ++v) {
          float s  = acc[v] + red[wIn & 3][lane][v];
          float x  = s + emb2[v] + c_bn;
          float h  = tanh_f(x);
          float z  = zb[v];
          float Hn = z * h + (1.0f - z) * hold2[v];
          Hf32[(c_mb + v) * HIDDEN + c_n] = Hn;
          Hout[(c_mb + v) * HIDDEN + c_n] = (_Float16)Hn;
        }
      }
    }
    grid_sync(bcnt, bgen, NWG);
  }
}

// ---------------------------------------------------------------------------
// Phase B: Y = H_all(16384x1024) @ Wo(1024x4096) + bo.  16x64 per wave.
__global__ void outproj_kernel(const _Float16* __restrict__ Hall,
                               const _Float16* __restrict__ Wot,
                               const float* __restrict__ bo,
                               float* __restrict__ Y)
{
  int wave = (int)((blockIdx.x * (size_t)blockDim.x + threadIdx.x) >> 5);
  int lane = threadIdx.x & 31;
  int row0 = (wave >> 6) * 16;
  int col0 = (wave & 63) * 64;

  v8f acc0 = {}, acc1 = {}, acc2 = {}, acc3 = {};
#pragma unroll 2
  for (int kc = 0; kc < HIDDEN; kc += 32) {
    v16h a  = load_a_frag(Hall, HIDDEN, row0, kc, lane);
    v16h b0 = load_b_frag(Wot, HIDDEN, col0 +  0, kc, lane);
    v16h b1 = load_b_frag(Wot, HIDDEN, col0 + 16, kc, lane);
    v16h b2 = load_b_frag(Wot, HIDDEN, col0 + 32, kc, lane);
    v16h b3 = load_b_frag(Wot, HIDDEN, col0 + 48, kc, lane);
    acc0 = __builtin_amdgcn_wmma_f32_16x16x32_f16(false, a, false, b0, (short)0, acc0, false, false);
    acc1 = __builtin_amdgcn_wmma_f32_16x16x32_f16(false, a, false, b1, (short)0, acc1, false, false);
    acc2 = __builtin_amdgcn_wmma_f32_16x16x32_f16(false, a, false, b2, (short)0, acc2, false, false);
    acc3 = __builtin_amdgcn_wmma_f32_16x16x32_f16(false, a, false, b3, (short)0, acc3, false, false);
  }

  int mb = row0 + ((lane & 16) ? 8 : 0);
#pragma unroll
  for (int j = 0; j < 4; ++j) {
    int n = col0 + j * 16 + (lane & 15);
    const v8f& acc = (j == 0) ? acc0 : (j == 1) ? acc1 : (j == 2) ? acc2 : acc3;
    float bn = bo[n];
#pragma unroll
    for (int v = 0; v < 8; ++v) {
      int m = mb + v;
      Y[(size_t)m * VOCAB + n] = acc[v] + bn;
    }
  }
}

__global__ void copy_hfinal_kernel(const float* __restrict__ Hf32, float* __restrict__ out) {
  int i = blockIdx.x * blockDim.x + threadIdx.x;
  if (i < BATCH * HIDDEN) out[i] = Hf32[i];
}

// ---------------------------------------------------------------------------
extern "C" void kernel_launch(void* const* d_in, const int* in_sizes, int n_in,
                              void* d_out, int out_size, void* d_ws, size_t ws_size,
                              hipStream_t stream) {
  const int*   X  = (const int*)  d_in[0];
  const float* Wz = (const float*)d_in[1];
  const float* bz = (const float*)d_in[2];
  const float* Wr = (const float*)d_in[3];
  const float* br = (const float*)d_in[4];
  const float* Wh = (const float*)d_in[5];
  const float* bh = (const float*)d_in[6];
  const float* Wo = (const float*)d_in[7];
  const float* bo = (const float*)d_in[8];

  // Workspace layout (~46.8 MiB total)
  char* p = (char*)d_ws;
  _Float16* Wzt   = (_Float16*)p; p += (size_t)HIDDEN * HIDDEN * sizeof(_Float16);
  _Float16* Wrt   = (_Float16*)p; p += (size_t)HIDDEN * HIDDEN * sizeof(_Float16);
  _Float16* Wht   = (_Float16*)p; p += (size_t)HIDDEN * HIDDEN * sizeof(_Float16);
  _Float16* Wot   = (_Float16*)p; p += (size_t)VOCAB  * HIDDEN * sizeof(_Float16);
  _Float16* Hall  = (_Float16*)p; p += (size_t)SEQ * BATCH * HIDDEN * sizeof(_Float16);
  _Float16* Hzero = (_Float16*)p; p += (size_t)BATCH * HIDDEN * sizeof(_Float16);
  float*    Hf32  = (float*)   p; p += (size_t)BATCH * HIDDEN * sizeof(float);
  float*    zbuf  = (float*)   p; p += (size_t)BATCH * HIDDEN * sizeof(float);
  _Float16* rH    = (_Float16*)p; p += (size_t)BATCH * HIDDEN * sizeof(_Float16);
  unsigned* bar   = (unsigned*)p; p += 2 * sizeof(unsigned);

  {
    size_t nW = (size_t)HIDDEN * HIDDEN;
    int blocks = (int)((nW + 255) / 256);
    convert_transpose_kernel<<<blocks, 256, 0, stream>>>(Wz, Wzt, HIDDEN, HIDDEN, VOCAB, HIDDEN);
    convert_transpose_kernel<<<blocks, 256, 0, stream>>>(Wr, Wrt, HIDDEN, HIDDEN, VOCAB, HIDDEN);
    convert_transpose_kernel<<<blocks, 256, 0, stream>>>(Wh, Wht, HIDDEN, HIDDEN, VOCAB, HIDDEN);
    size_t nO = (size_t)VOCAB * HIDDEN;
    convert_transpose_kernel<<<(int)((nO + 255) / 256), 256, 0, stream>>>(Wo, Wot, HIDDEN, VOCAB, 0, VOCAB);
  }
  init_kernel<<<(BATCH * HIDDEN + 255) / 256, 256, 0, stream>>>(Hf32, Hzero, bar);

  gru_scan_kernel<<<NWG, 256, 0, stream>>>(Wzt, Wrt, Wht, Wz, Wr, Wh,
                                           bz, br, bh, X,
                                           Hzero, Hf32, zbuf, rH, Hall, bar);

  outproj_kernel<<<8192, 256, 0, stream>>>(Hall, Wot, bo, (float*)d_out);

  copy_hfinal_kernel<<<(BATCH * HIDDEN + 255) / 256, 256, 0, stream>>>(
      Hf32, (float*)d_out + (size_t)SEQ * BATCH * VOCAB);
}
